// FormulaParser_61804579390123
// MI455X (gfx1250) — compile-verified, matrix operand order checked by reading, not run on hardware
//
#include <hip/hip_runtime.h>
#include <cstddef>

// ---------------- problem constants ----------------
#define TT   8192
#define HH   1024
#define II   1024
#define G3   3072    // 3*H
#define OO   2048    // 2*H
#define LLMAX 64

typedef __attribute__((ext_vector_type(16))) __bf16 v16bf;
typedef __attribute__((ext_vector_type(8)))  float  v8f;
typedef __attribute__((ext_vector_type(4)))  unsigned int v4u;
typedef __attribute__((ext_vector_type(8)))  int v8i;
typedef __attribute__((ext_vector_type(4)))  int v4i;

struct bfrag { uint4 lo, hi; };   // 32B == v16bf

#ifndef __has_builtin
#define __has_builtin(x) 0
#endif

#if defined(__AMDGCN__) && __has_builtin(__builtin_amdgcn_tensor_load_to_lds) && \
    __has_builtin(__builtin_amdgcn_s_wait_tensorcnt)
#define USE_TDM 1
#endif

__device__ __forceinline__ float sigm(float x) { return 1.0f / (1.0f + __expf(-x)); }

// ---------------- fp32 -> bf16 (RNE) ----------------
__global__ void k_cvt_bf16(const float* __restrict__ s, unsigned short* __restrict__ d, int n) {
    for (int i = blockIdx.x * blockDim.x + threadIdx.x; i < n; i += gridDim.x * blockDim.x) {
        unsigned u = __float_as_uint(s[i]);
        unsigned r = u + 0x7FFFu + ((u >> 16) & 1u);
        d[i] = (unsigned short)(r >> 16);
    }
}

#ifdef USE_TDM
// Build a 2D TDM descriptor (D#) and DMA a (64 rows x 32 cols) bf16 tile,
// row-major, from a (nrows x II) bf16 tensor into LDS at lds_addr.
// D# layout per CDNA5 ISA ch.8: group0 = {count/lds_addr/global_addr/type},
// group1 = {data_size, tensor dims, tile dims, dim0 stride}.
// This toolchain's builtin is the 6-arg form:
//   (uint32x4 g0, int32x8 g1, int32x4 g2, int32x4 g3, int32x8 extra, i32 cpol)
__device__ __forceinline__ void tdm_tile_load(unsigned lds_addr,
                                              const unsigned short* g,
                                              unsigned nrows, unsigned row0,
                                              unsigned kc)
{
    unsigned long long addr =
        (unsigned long long)(size_t)g + ((unsigned long long)row0 * II + kc) * 2ull;
    v4u g0;
    g0[0] = 1u;                                               // count = 1 valid descriptor
    g0[1] = lds_addr;                                         // LDS byte address
    g0[2] = (unsigned)addr;                                   // global addr [31:0]
    g0[3] = ((unsigned)(addr >> 32) & 0x01FFFFFFu)            // global addr [56:32]
            | 0x80000000u;                                    // type = 2 ("image")
    v8i g1;
    g1[0] = 0x00010000;                                       // data_size = 2 bytes
    g1[1] = (int)(((unsigned)II & 0xFFFFu) << 16);            // tensor_dim0 lo16
    g1[2] = (int)((((unsigned)II >> 16) & 0xFFFFu)
            | ((nrows & 0xFFFFu) << 16));                     // dim0 hi16 | tensor_dim1 lo16
    g1[3] = (int)(((nrows >> 16) & 0xFFFFu) | (32u << 16));   // dim1 hi16 | tile_dim0 = 32
    g1[4] = 64;                                               // tile_dim1 = 64 rows
    g1[5] = (int)II;                                          // tensor_dim0_stride
    g1[6] = 0;
    g1[7] = 0;
    v4i z4 = {0, 0, 0, 0};                                    // groups 2/3 unused (2D)
    v8i z8 = {0, 0, 0, 0, 0, 0, 0, 0};
    __builtin_amdgcn_tensor_load_to_lds(g0, g1, z4, z4, z8, 0);
}
#endif

// Assemble ISA 7.12.2 16-bit fragments from an LDS row-major 64x32 tile and
// issue the two WMMAs for this wave's 16(M) x 32(N) sub-tile.
__device__ __forceinline__ void frag_mma(const unsigned short* __restrict__ xs,
                                         const unsigned short* __restrict__ ws,
                                         int m_sub, int n_sub, int l16, int hlf,
                                         v8f& acc0, v8f& acc1)
{
    bfrag fa, fb0, fb1;
    const unsigned short* ap = xs + (m_sub + l16) * 32;
    fa.lo = *(const uint4*)(ap + 8 * hlf);
    fa.hi = *(const uint4*)(ap + 16 + 8 * hlf);
    const unsigned short* bp0 = ws + (n_sub + l16) * 32;
    fb0.lo = *(const uint4*)(bp0 + 8 * hlf);
    fb0.hi = *(const uint4*)(bp0 + 16 + 8 * hlf);
    const unsigned short* bp1 = ws + (n_sub + 16 + l16) * 32;
    fb1.lo = *(const uint4*)(bp1 + 8 * hlf);
    fb1.hi = *(const uint4*)(bp1 + 16 + 8 * hlf);
    v16bf a  = __builtin_bit_cast(v16bf, fa);
    v16bf b0 = __builtin_bit_cast(v16bf, fb0);
    v16bf b1 = __builtin_bit_cast(v16bf, fb1);
    acc0 = __builtin_amdgcn_wmma_f32_16x16x32_bf16(false, a, false, b0, (short)0, acc0, false, false);
    acc1 = __builtin_amdgcn_wmma_f32_16x16x32_bf16(false, a, false, b1, (short)0, acc1, false, false);
}

// ---------------- WMMA GEMM: G[dir] = x @ Wih[dir]^T + bih[dir] ----------------
// C is (T x 3H), A = x bf16 (T x I), B = Wih bf16 (3H x I) used transposed.
// Block tile 64(M) x 64(N); 8 waves; TDM double-buffered LDS pipeline.
__global__ __launch_bounds__(256) void k_gemm_wmma(
    const unsigned short* __restrict__ Xb,
    const unsigned short* __restrict__ Wfb,
    const unsigned short* __restrict__ Wbb,
    const float* __restrict__ bih_f, const float* __restrict__ bih_b,
    float* __restrict__ Gf, float* __restrict__ Gb)
{
    __shared__ alignas(16) unsigned short sX[2][64 * 32];
    __shared__ alignas(16) unsigned short sW[2][64 * 32];

    const int tid  = threadIdx.x;
    const int wv   = tid >> 5;
    const int lane = tid & 31;
    const int hlf  = lane >> 4;      // K interleave half
    const int l16  = lane & 15;

    const int m0  = blockIdx.y * 64;
    const int n0  = blockIdx.x * 64;
    const int dir = blockIdx.z;

    const unsigned short* W = dir ? Wbb : Wfb;
    const float* bias       = dir ? bih_b : bih_f;
    float* G                = dir ? Gb : Gf;

    const int m_sub = (wv & 3) * 16;
    const int n_sub = (wv >> 2) * 32;

    v8f acc0 = {0.f,0.f,0.f,0.f,0.f,0.f,0.f,0.f};
    v8f acc1 = {0.f,0.f,0.f,0.f,0.f,0.f,0.f,0.f};

#ifdef USE_TDM
    const unsigned ldsX0 = (unsigned)(size_t)&sX[0][0];
    const unsigned ldsX1 = (unsigned)(size_t)&sX[1][0];
    const unsigned ldsW0 = (unsigned)(size_t)&sW[0][0];
    const unsigned ldsW1 = (unsigned)(size_t)&sW[1][0];
    // prologue: DMA first tiles into buffer 0 (wave 0 drives the TDM)
    if (wv == 0) {
        tdm_tile_load(ldsX0, Xb, TT, (unsigned)m0, 0u);
        tdm_tile_load(ldsW0, W,  G3, (unsigned)n0, 0u);
        __builtin_amdgcn_s_wait_tensorcnt(0);
    }
    __syncthreads();
    int cur = 0;
    for (int kc = 0; kc < II; kc += 32) {
        const bool more = (kc + 32) < II;
        if (wv == 0 && more) {   // async prefetch of next K-chunk into other buffer
            tdm_tile_load(cur ? ldsX0 : ldsX1, Xb, TT, (unsigned)m0, (unsigned)(kc + 32));
            tdm_tile_load(cur ? ldsW0 : ldsW1, W,  G3, (unsigned)n0, (unsigned)(kc + 32));
        }
        frag_mma(sX[cur], sW[cur], m_sub, n_sub, l16, hlf, acc0, acc1);
        if (wv == 0 && more) __builtin_amdgcn_s_wait_tensorcnt(0);
        __syncthreads();
        cur ^= 1;
    }
#else
    // fallback: manual global->LDS staging, single buffer
    const int lr  = tid >> 2;
    const int seg = tid & 3;
    for (int kc = 0; kc < II; kc += 32) {
        *(uint4*)&sX[0][lr * 32 + seg * 8] =
            *(const uint4*)(Xb + (size_t)(m0 + lr) * II + kc + seg * 8);
        *(uint4*)&sW[0][lr * 32 + seg * 8] =
            *(const uint4*)(W  + (size_t)(n0 + lr) * II + kc + seg * 8);
        __syncthreads();
        frag_mma(sX[0], sW[0], m_sub, n_sub, l16, hlf, acc0, acc1);
        __syncthreads();
    }
#endif

    // C/D layout: elem r -> row = r + 8*half, col = lane&15
    const int col0 = n0 + n_sub + l16;
    const int col1 = col0 + 16;
    const float bv0 = bias[col0];
    const float bv1 = bias[col1];
#pragma unroll
    for (int r = 0; r < 8; ++r) {
        const int row = m0 + m_sub + r + 8 * hlf;
        G[(size_t)row * G3 + col0] = acc0[r] + bv0;
        G[(size_t)row * G3 + col1] = acc1[r] + bv1;
    }
}

// ---------------- persistent recurrent GRU scan (1 WG per direction) ----------------
__global__ __launch_bounds__(1024) void k_scan(
    const float* __restrict__ h0,
    const float* __restrict__ Whh_f, const float* __restrict__ Whh_b,
    const float* __restrict__ bhh_f, const float* __restrict__ bhh_b,
    const float* __restrict__ Gf,    const float* __restrict__ Gb,
    float* __restrict__ out_f, float* __restrict__ out_b,
    float* __restrict__ h_dec)
{
    const int dir = blockIdx.x;
    const int i   = threadIdx.x;

    const float* Whh = dir ? Whh_b : Whh_f;
    const float* bhh = dir ? bhh_b : bhh_f;
    const float* G   = dir ? Gb : Gf;
    float* outp      = dir ? out_b : out_f;

    __shared__ float h[HH];
    h[i] = h0[dir * HH + i];
    __syncthreads();

    const float* r0 = Whh + (size_t)i * HH;
    const float* r1 = Whh + (size_t)(HH + i) * HH;
    const float* r2 = Whh + (size_t)(2 * HH + i) * HH;
    const float b_r = bhh[i], b_z = bhh[HH + i], b_n = bhh[2 * HH + i];

    for (int t = 0; t < TT; ++t) {
        const int idx = dir ? (TT - 1 - t) : t;
        if (t + 1 < TT) {       // prefetch next step's gi rows (L2-resident stream)
            const int nidx = dir ? (TT - 2 - t) : (t + 1);
            const float* gp = G + (size_t)nidx * G3;
            __builtin_prefetch(gp + i, 0, 1);
            __builtin_prefetch(gp + HH + i, 0, 1);
            __builtin_prefetch(gp + 2 * HH + i, 0, 1);
        }
        float hr = b_r, hz = b_z, hn = b_n;
        for (int k = 0; k < HH; k += 4) {
            const float4 hv = *(const float4*)&h[k];
            const float4 w0 = *(const float4*)(r0 + k);
            const float4 w1 = *(const float4*)(r1 + k);
            const float4 w2 = *(const float4*)(r2 + k);
            hr += hv.x * w0.x + hv.y * w0.y + hv.z * w0.z + hv.w * w0.w;
            hz += hv.x * w1.x + hv.y * w1.y + hv.z * w1.z + hv.w * w1.w;
            hn += hv.x * w2.x + hv.y * w2.y + hv.z * w2.z + hv.w * w2.w;
        }
        const float* g = G + (size_t)idx * G3;
        const float gr = g[i], gz = g[HH + i], gn = g[2 * HH + i];
        const float rr = sigm(gr + hr);
        const float zz = sigm(gz + hz);
        const float nn = tanhf(gn + rr * hn);
        const float hnew = (1.0f - zz) * nn + zz * h[i];
        __syncthreads();
        h[i] = hnew;
        outp[(size_t)idx * HH + i] = hnew;
        __syncthreads();
    }
    if (dir == 1) h_dec[i] = h[i];   // final backward hidden -> decoder h0
}

// ---------------- enc_scores[t] = enc2[t] . W_att_e ----------------
__global__ __launch_bounds__(256) void k_scores(
    const float* __restrict__ out_f, const float* __restrict__ out_b,
    const float* __restrict__ W_att, float* __restrict__ sc)
{
    const int t = blockIdx.x * 256 + threadIdx.x;
    if (t >= TT) return;
    const float* wf = W_att + HH;
    const float* wb = W_att + 2 * HH;
    const float* pf = out_f + (size_t)t * HH;
    const float* pb = out_b + (size_t)t * HH;
    float s = 0.f;
    for (int k = 0; k < HH; k += 4) {
        const float4 a = *(const float4*)(pf + k), w = *(const float4*)(wf + k);
        const float4 b = *(const float4*)(pb + k), v = *(const float4*)(wb + k);
        s += a.x * w.x + a.y * w.y + a.z * w.z + a.w * w.w;
        s += b.x * v.x + b.y * v.y + b.z * v.z + b.w * v.w;
    }
    sc[t] = s;
}

// ---------------- block-wide reductions (1024 threads) ----------------
__device__ __forceinline__ float block_sum(float* sred, float v) {
    const int tid = threadIdx.x;
    sred[tid] = v; __syncthreads();
    for (int off = 512; off > 0; off >>= 1) {
        if (tid < off) sred[tid] += sred[tid + off];
        __syncthreads();
    }
    const float r = sred[0]; __syncthreads(); return r;
}
__device__ __forceinline__ float block_max(float* sred, float v) {
    const int tid = threadIdx.x;
    sred[tid] = v; __syncthreads();
    for (int off = 512; off > 0; off >>= 1) {
        if (tid < off) sred[tid] = fmaxf(sred[tid], sred[tid + off]);
        __syncthreads();
    }
    const float r = sred[0]; __syncthreads(); return r;
}

// ---------------- persistent attention decoder (1 WG, 64 serial steps) ----------------
__global__ __launch_bounds__(1024) void k_decoder(
    const float* __restrict__ out_f, const float* __restrict__ out_b,
    const float* __restrict__ enc_sc, const float* __restrict__ h_dec0,
    const float* __restrict__ dWih, const float* __restrict__ dWhh,
    const float* __restrict__ dbih, const float* __restrict__ dbhh,
    const float* __restrict__ W_att, const float* __restrict__ b_att,
    const float* __restrict__ W_out, const float* __restrict__ b_out,
    const int* __restrict__ max_len, float* __restrict__ out)
{
    const int tid = threadIdx.x;
    __shared__ float sh[HH];
    __shared__ float shn[HH];
    __shared__ float sxin[4 * HH];
    __shared__ float sattn[TT];
    __shared__ float sred[1024];

    int ml = max_len[0];
    if (ml > LLMAX) ml = LLMAX;
    if (ml < 0) ml = 0;

    sh[tid] = h_dec0[tid];
    __syncthreads();

    for (int s = 0; s < ml; ++s) {
        // attention logits: enc_scores[t] + h.W_att_h + b_att
        const float hw  = block_sum(sred, sh[tid] * W_att[tid]);
        const float sc0 = hw + b_att[0];
        float lm = -3.4e38f;
        for (int t = tid; t < TT; t += 1024) {
            const float v = enc_sc[t] + sc0;
            sattn[t] = v;
            lm = fmaxf(lm, v);
        }
        const float mx = block_max(sred, lm);
        float ls = 0.f;
        for (int t = tid; t < TT; t += 1024) {
            const float e = __expf(sattn[t] - mx);
            sattn[t] = e;
            ls += e;
        }
        const float Z = block_sum(sred, ls);
        const float inv = 1.0f / Z;

        // context = softmax . enc2  (coalesced: thread = column)
        float ca = 0.f, cb = 0.f;
        for (int t = 0; t < TT; ++t) {
            const float a = sattn[t];
            ca += a * out_f[(size_t)t * HH + tid];
            cb += a * out_b[(size_t)t * HH + tid];
        }
        sxin[2 * HH + tid] = ca * inv;
        sxin[3 * HH + tid] = cb * inv;
        // dec_in = previous log-softmax output (zeros for step 0)
        if (s == 0) {
            sxin[tid] = 0.f; sxin[HH + tid] = 0.f;
        } else {
            sxin[tid]       = out[(size_t)(s - 1) * OO + tid];
            sxin[HH + tid]  = out[(size_t)(s - 1) * OO + HH + tid];
        }
        __syncthreads();

        // GRU cell: gi = xin @ dWih^T + dbih ; gh = h @ dWhh^T + dbhh
        float gr = dbih[tid], gz = dbih[HH + tid], gn = dbih[2 * HH + tid];
        {
            const float* wr = dWih + (size_t)tid * (4 * HH);
            const float* wz = dWih + (size_t)(HH + tid) * (4 * HH);
            const float* wn = dWih + (size_t)(2 * HH + tid) * (4 * HH);
            for (int k = 0; k < 4 * HH; k += 4) {
                const float4 xv = *(const float4*)&sxin[k];
                const float4 a = *(const float4*)(wr + k);
                const float4 b = *(const float4*)(wz + k);
                const float4 c = *(const float4*)(wn + k);
                gr += xv.x * a.x + xv.y * a.y + xv.z * a.z + xv.w * a.w;
                gz += xv.x * b.x + xv.y * b.y + xv.z * b.z + xv.w * b.w;
                gn += xv.x * c.x + xv.y * c.y + xv.z * c.z + xv.w * c.w;
            }
        }
        float hr = dbhh[tid], hz = dbhh[HH + tid], hn = dbhh[2 * HH + tid];
        {
            const float* wr = dWhh + (size_t)tid * HH;
            const float* wz = dWhh + (size_t)(HH + tid) * HH;
            const float* wn = dWhh + (size_t)(2 * HH + tid) * HH;
            for (int k = 0; k < HH; k += 4) {
                const float4 hv = *(const float4*)&sh[k];
                const float4 a = *(const float4*)(wr + k);
                const float4 b = *(const float4*)(wz + k);
                const float4 c = *(const float4*)(wn + k);
                hr += hv.x * a.x + hv.y * a.y + hv.z * a.z + hv.w * a.w;
                hz += hv.x * b.x + hv.y * b.y + hv.z * b.z + hv.w * b.w;
                hn += hv.x * c.x + hv.y * c.y + hv.z * c.z + hv.w * c.w;
            }
        }
        const float rr = sigm(gr + hr);
        const float zz = sigm(gz + hz);
        const float nn = tanhf(gn + rr * hn);
        const float hnew = (1.0f - zz) * nn + zz * sh[tid];
        shn[tid] = hnew;
        __syncthreads();

        // logits + log_softmax over 2048 (thread owns cols tid, tid+1024)
        float l0 = b_out[tid], l1 = b_out[HH + tid];
        {
            const float* o0 = W_out + (size_t)tid * HH;
            const float* o1 = W_out + (size_t)(HH + tid) * HH;
            for (int k = 0; k < HH; k += 4) {
                const float4 hv = *(const float4*)&shn[k];
                const float4 a = *(const float4*)(o0 + k);
                const float4 b = *(const float4*)(o1 + k);
                l0 += hv.x * a.x + hv.y * a.y + hv.z * a.z + hv.w * a.w;
                l1 += hv.x * b.x + hv.y * b.y + hv.z * b.z + hv.w * b.w;
            }
        }
        const float m2   = block_max(sred, fmaxf(l0, l1));
        const float zsum = block_sum(sred, __expf(l0 - m2) + __expf(l1 - m2));
        const float lse  = m2 + __logf(zsum);
        out[(size_t)s * OO + tid]      = l0 - lse;
        out[(size_t)s * OO + HH + tid] = l1 - lse;
        __syncthreads();
        sh[tid] = hnew;
        __syncthreads();
    }
}

// ---------------- host launch ----------------
extern "C" void kernel_launch(void* const* d_in, const int* in_sizes, int n_in,
                              void* d_out, int out_size, void* d_ws, size_t ws_size,
                              hipStream_t stream)
{
    (void)in_sizes; (void)n_in; (void)out_size; (void)ws_size;

    const float* x        = (const float*)d_in[0];
    const float* h0       = (const float*)d_in[1];
    const float* eWih_f   = (const float*)d_in[2];
    const float* eWhh_f   = (const float*)d_in[3];
    const float* ebih_f   = (const float*)d_in[4];
    const float* ebhh_f   = (const float*)d_in[5];
    const float* eWih_b   = (const float*)d_in[6];
    const float* eWhh_b   = (const float*)d_in[7];
    const float* ebih_b   = (const float*)d_in[8];
    const float* ebhh_b   = (const float*)d_in[9];
    const float* dWih     = (const float*)d_in[10];
    const float* dWhh     = (const float*)d_in[11];
    const float* dbih     = (const float*)d_in[12];
    const float* dbhh     = (const float*)d_in[13];
    const float* W_att    = (const float*)d_in[14];
    const float* b_att    = (const float*)d_in[15];
    const float* W_out    = (const float*)d_in[16];
    const float* b_out    = (const float*)d_in[17];
    const int*   max_len  = (const int*)d_in[18];

    // workspace partition (256B aligned slices)
    char* base = (char*)d_ws;
    size_t off = 0;
    auto take = [&](size_t bytes) -> char* {
        char* p = base + off;
        off = (off + bytes + 255) & ~(size_t)255;
        return p;
    };
    unsigned short* xb   = (unsigned short*)take((size_t)TT * II * 2);
    unsigned short* wfb  = (unsigned short*)take((size_t)G3 * II * 2);
    unsigned short* wbb  = (unsigned short*)take((size_t)G3 * II * 2);
    float* Gf     = (float*)take((size_t)TT * G3 * 4);
    float* Gb     = (float*)take((size_t)TT * G3 * 4);
    float* out_f  = (float*)take((size_t)TT * HH * 4);
    float* out_b  = (float*)take((size_t)TT * HH * 4);
    float* enc_sc = (float*)take((size_t)TT * 4);
    float* h_dec  = (float*)take((size_t)HH * 4);

    // 1) bf16 conversions for WMMA operands
    k_cvt_bf16<<<dim3(2048), dim3(256), 0, stream>>>(x,      xb,  TT * II);
    k_cvt_bf16<<<dim3(2048), dim3(256), 0, stream>>>(eWih_f, wfb, G3 * II);
    k_cvt_bf16<<<dim3(2048), dim3(256), 0, stream>>>(eWih_b, wbb, G3 * II);

    // 2) input projections for both directions (TDM-fed WMMA GEMMs, ~103 GFLOP)
    k_gemm_wmma<<<dim3(G3 / 64, TT / 64, 2), dim3(256), 0, stream>>>(
        xb, wfb, wbb, ebih_f, ebih_b, Gf, Gb);

    // 3) serial GRU scans (forward + backward concurrently)
    k_scan<<<dim3(2), dim3(1024), 0, stream>>>(
        h0, eWhh_f, eWhh_b, ebhh_f, ebhh_b, Gf, Gb, out_f, out_b, h_dec);

    // 4) attention score precompute
    k_scores<<<dim3(TT / 256), dim3(256), 0, stream>>>(out_f, out_b, W_att, enc_sc);

    // 5) persistent attention decoder
    k_decoder<<<dim3(1), dim3(1024), 0, stream>>>(
        out_f, out_b, enc_sc, h_dec, dWih, dWhh, dbih, dbhh,
        W_att, b_att, W_out, b_out, max_len, (float*)d_out);
}
